// EdgeMLP_47665547051556
// MI455X (gfx1250) — matrix operand order repaired and checked by reading, not compiled, not run on hardware
//
#include <hip/hip_runtime.h>
#include <hip/hip_bf16.h>

typedef __attribute__((ext_vector_type(16))) _Float16 v16h;
typedef __attribute__((ext_vector_type(8)))  float    v8f;

__global__ void edge_mlp_zero_kernel(float* __restrict__ f, int n) {
    int i = blockIdx.x * blockDim.x + threadIdx.x;
    if (i < n) f[i] = 0.0f;
}

__global__ __launch_bounds__(256) void EdgeMLP_wmma_kernel(
    const float* __restrict__ x,
    const int*   __restrict__ ei,     // [2, E] int32: row0 = src, row1 = dst
    const float* __restrict__ u,
    const float* __restrict__ W1,     // [2, 32]
    const float* __restrict__ b1,     // [32]
    const float* __restrict__ W2,     // [32, 32]
    const float* __restrict__ b2,     // [32]
    const float* __restrict__ W3,     // [32, 1]
    const float* __restrict__ b3,     // [1]
    float* __restrict__ f,
    int nEdges)
{
    const int lane   = threadIdx.x & 31;
    const int wave   = (blockIdx.x * blockDim.x + threadIdx.x) >> 5;
    const int nWaves = (gridDim.x * blockDim.x) >> 5;
    const int half   = lane >> 4;      // 0: lanes 0-15, 1: lanes 16-31
    const int n0     = lane & 15;      // WMMA column index == edge slot in tile

    // ============ Per-wave constants (hoisted out of the edge loop) ============
    // Transposed formulation: D[feat, edge] = A(=W2^T block) x B(=H1^T) + C(=b2)
    //
    // A fragment (16x32 f16): lane holds row M = n0; element i -> K = kA(i).
    const int kshift = half * 8;
    v16h a0, a1;   // A0 = W2^T rows 0-15 (out-features 0-15); A1 = rows 16-31
    #pragma unroll
    for (int i = 0; i < 16; ++i) {
        const int kA = kshift + (i < 8 ? i : i + 8);   // hidden-input index
        a0[i] = (_Float16)W2[kA * 32 + n0];            // W2^T[n0][kA]
        a1[i] = (_Float16)W2[kA * 32 + 16 + n0];       // W2^T[16+n0][kA]
    }
    // C fragments carry the layer-2 bias: C[M,N] = b2[M]; D row r -> M = r + 8*half.
    v8f cb0, cb1;
    float w3v0[8], w3v1[8];
    #pragma unroll
    for (int r = 0; r < 8; ++r) {
        const int m = r + 8 * half;
        cb0[r] = b2[m];          w3v0[r] = W3[m];        // features 0..15
        cb1[r] = b2[16 + m];     w3v1[r] = W3[16 + m];   // features 16..31
    }
    // Layer-1 weights for the B (=H1^T) build: element i -> K = 8*half + (i < 8 ? i : i + 8).
    float w1s[16], w1d[16], b1v[16];
    #pragma unroll
    for (int i = 0; i < 16; ++i) {
        const int kB = half * 8 + (i < 8 ? i : i + 8);
        w1s[i] = W1[kB];          // W1[0][kB]
        w1d[i] = W1[32 + kB];     // W1[1][kB]
        b1v[i] = b1[kB];
    }
    const float halfB3 = 0.5f * b3[0];   // b3 folded into the two half-sums

    // ============ Main loop: 32 edges (two 16-edge tiles) per iteration ============
    const int nPairs = (nEdges + 31) >> 5;
    for (int pair = wave; pair < nPairs; pair += nWaves) {
        const int eBase = pair << 5;
        const int e     = eBase + lane;
        const bool haveEdge = (e < nEdges);

        // Coalesced 128B index loads + L2-resident gathers (all 32 lanes).
        int   srcIdx = 0;
        float xs_l = 0.0f, xd_l = 0.0f, uv_l = 0.0f;
        if (haveEdge) {
            srcIdx      = ei[e];
            const int d = ei[nEdges + e];
            xs_l = x[srcIdx];
            xd_l = x[d];
            uv_l = u[d];
        }
        // Each lane needs the scalars of ITS column's edge:
        //   tile0: edge eBase + n0   (owners: lanes 0-15)
        //   tile1: edge eBase+16+n0  (owners: lanes 16-31)
        const float xs0 = __shfl(xs_l, n0, 32);
        const float xd0 = __shfl(xd_l, n0, 32);
        const float xs1 = __shfl(xs_l, 16 + n0, 32);
        const float xd1 = __shfl(xd_l, 16 + n0, 32);

        // Layer 1 fused into the B (=H1^T) fragment build.
        v16h bf0, bf1;
        #pragma unroll
        for (int i = 0; i < 16; ++i) {
            float h0 = fmaf(xs0, w1s[i], fmaf(xd0, w1d[i], b1v[i]));
            float h1 = fmaf(xs1, w1s[i], fmaf(xd1, w1d[i], b1v[i]));
            bf0[i] = (_Float16)fmaxf(h0, 0.0f);
            bf1[i] = (_Float16)fmaxf(h1, 0.0f);
        }

        // Layer 2 (+bias via C): four 16x16x32 WMMAs.
        // d<t><g>: tile t, feature-group g; D[r] = h2[edge n0][feat r+8*half(+16g)]
        v8f d00 = __builtin_amdgcn_wmma_f32_16x16x32_f16(
                      false, a0, false, bf0, (short)0, cb0, false, false);
        v8f d01 = __builtin_amdgcn_wmma_f32_16x16x32_f16(
                      false, a1, false, bf0, (short)0, cb1, false, false);
        v8f d10 = __builtin_amdgcn_wmma_f32_16x16x32_f16(
                      false, a0, false, bf1, (short)0, cb0, false, false);
        v8f d11 = __builtin_amdgcn_wmma_f32_16x16x32_f16(
                      false, a1, false, bf1, (short)0, cb1, false, false);

        // Layer 3: per-lane partial sum over this half's 16 features.
        float acc0 = halfB3, acc1 = halfB3;
        #pragma unroll
        for (int r = 0; r < 8; ++r) {
            acc0 = fmaf(fmaxf(d00[r], 0.0f), w3v0[r], acc0);
            acc0 = fmaf(fmaxf(d01[r], 0.0f), w3v1[r], acc0);
            acc1 = fmaf(fmaxf(d10[r], 0.0f), w3v0[r], acc1);
            acc1 = fmaf(fmaxf(d11[r], 0.0f), w3v1[r], acc1);
        }
        // Merge the two feature halves: lanes l and l^16 hold the same edge.
        acc0 += __shfl_xor(acc0, 16, 32);
        acc1 += __shfl_xor(acc1, 16, 32);

        // Owner lanes already hold their edge weight (b3 included): no routing.
        const float we = (lane < 16) ? acc0 : acc1;

        // Scatter: f[src] += w_e * u[dst]  (one atomic per edge/lane)
        if (haveEdge) {
            atomicAdd(&f[srcIdx], we * uv_l);
        }
    }
}

extern "C" void kernel_launch(void* const* d_in, const int* in_sizes, int n_in,
                              void* d_out, int out_size, void* d_ws, size_t ws_size,
                              hipStream_t stream) {
    const float* x  = (const float*)d_in[0];
    const int*   ei = (const int*)  d_in[1];
    const float* u  = (const float*)d_in[2];
    const float* W1 = (const float*)d_in[3];
    const float* b1 = (const float*)d_in[4];
    const float* W2 = (const float*)d_in[5];
    const float* b2 = (const float*)d_in[6];
    const float* W3 = (const float*)d_in[7];
    const float* b3 = (const float*)d_in[8];
    float* f = (float*)d_out;

    const int nNodes = in_sizes[0];
    const int nEdges = in_sizes[1] / 2;

    // Zero the output (harness poisons it; atomics accumulate into it).
    edge_mlp_zero_kernel<<<(nNodes + 255) / 256, 256, 0, stream>>>(f, nNodes);

    // 3072 blocks x 8 waves = 24576 waves; 200K 32-edge pairs -> ~8 iters/wave.
    const int blocks = 3072;
    EdgeMLP_wmma_kernel<<<blocks, 256, 0, stream>>>(
        x, ei, u, W1, b1, W2, b2, W3, b3, f, nEdges);
}
